// MLAAttention_50319836840187
// MI455X (gfx1250) — matrix operand order repaired and checked by reading, not compiled, not run on hardware
//
#include <hip/hip_runtime.h>
#include <hip/hip_bf16.h>

typedef __attribute__((ext_vector_type(16))) _Float16 v16h;
typedef __attribute__((ext_vector_type(8)))  _Float16 v8h;
typedef __attribute__((ext_vector_type(8)))  float    v8f;
typedef __attribute__((ext_vector_type(4)))  unsigned int v4u;
typedef __attribute__((ext_vector_type(4)))  int v4i;
typedef __attribute__((ext_vector_type(8)))  int v8i;

typedef __attribute__((address_space(1))) v4i* as1_v4i_ptr;
typedef __attribute__((address_space(3))) v4i* as3_v4i_ptr;

#define WMMA_F32_F16(a, b, c) \
  __builtin_amdgcn_wmma_f32_16x16x32_f16(false, (a), false, (b), (short)0, (c), false, false)

// -------------------------- CDNA5 feature probes ---------------------------
#if __has_builtin(__builtin_amdgcn_global_load_async_to_lds_b128)
#define HAVE_ASYNC 1
__device__ __forceinline__ void async_copy_b128(const void* g, void* l) {
  __builtin_amdgcn_global_load_async_to_lds_b128(
      (as1_v4i_ptr)(const void*)g, (as3_v4i_ptr)l, 0, 0);
}
__device__ __forceinline__ void wait_async0() {
#if __has_builtin(__builtin_amdgcn_s_wait_asynccnt)
  __builtin_amdgcn_s_wait_asynccnt(0);
#else
  asm volatile("s_wait_asynccnt 0" ::: "memory");
#endif
}
__device__ __forceinline__ void wait_async5() {
#if __has_builtin(__builtin_amdgcn_s_wait_asynccnt)
  __builtin_amdgcn_s_wait_asynccnt(5);
#else
  asm volatile("s_wait_asynccnt 5" ::: "memory");
#endif
}
#endif

#if __has_builtin(__builtin_amdgcn_tensor_load_to_lds)
#define HAVE_TDM 1
__device__ __forceinline__ unsigned lds_addr_of(const void* p) {
  return (unsigned)(unsigned long long)
      (__attribute__((address_space(3))) const char*)p;
}
__device__ __forceinline__ void tdm_load(v4u g0, v8i g1) {
  v4i z4 = {0, 0, 0, 0};
#if __has_include(<hip/amd_detail/amd_gfx1250_TDM.h>)
  v8i z8 = {0, 0, 0, 0, 0, 0, 0, 0};
  __builtin_amdgcn_tensor_load_to_lds(g0, g1, z4, z4, z8, 0);
#else
  __builtin_amdgcn_tensor_load_to_lds(g0, g1, z4, z4, 0);
#endif
}
__device__ __forceinline__ void wait_tensor0() {
#if __has_builtin(__builtin_amdgcn_s_wait_tensorcnt)
  __builtin_amdgcn_s_wait_tensorcnt(0);
#else
  asm volatile("s_wait_tensorcnt 0" ::: "memory");
#endif
}
__device__ __forceinline__ void wait_tensor2() {
#if __has_builtin(__builtin_amdgcn_s_wait_tensorcnt)
  __builtin_amdgcn_s_wait_tensorcnt(2);
#else
  asm volatile("s_wait_tensorcnt 2" ::: "memory");
#endif
}
// 2-D tile load: src is row-major [dim1][ld] halves; copies a [128][32]
// tile at (r0, k0) into an LDS buffer with 40-half row stride
// (pad_interval = 16 DWORDs, pad_amount = 4 DWORDs). OOB rows read zero.
__device__ __forceinline__ void tdm_issue_tile(const _Float16* src, int ld,
                                               int dim1, int r0, int k0,
                                               void* ldsbuf) {
  unsigned long long ga =
      (unsigned long long)(const void*)(src + (size_t)r0 * ld + k0);
  v4u g0;
  g0[0] = 1u;                                   // count=1, user mode
  g0[1] = lds_addr_of(ldsbuf);                  // lds_addr (bytes)
  g0[2] = (unsigned)(ga & 0xFFFFFFFFu);         // global_addr[31:0]
  g0[3] = ((unsigned)(ga >> 32) & 0x01FFFFFFu)  // global_addr[56:32]
          | (2u << 30);                         // type = 2 (image)
  v8i g1;
  g1[0] = (int)((1u << 16)      // data_size = 1 (2 bytes)
              | (1u << 20)      // pad_enable
              | (3u << 22)      // pad_interval: 16 DWORDs (= 32 halves)
              | (3u << 25));    // pad_amount: 4 DWORDs (= 8 halves)
  g1[1] = (int)(((unsigned)ld & 0xFFFFu) << 16);                   // dim0 lo
  g1[2] = (int)(((unsigned)ld >> 16) |
                (((unsigned)dim1 & 0xFFFFu) << 16));               // dim0 hi, dim1 lo
  g1[3] = (int)(((unsigned)dim1 >> 16) | (32u << 16));             // dim1 hi, tile_dim0=32
  g1[4] = (int)128u;                                               // tile_dim1=128
  g1[5] = (int)(unsigned)ld;                                       // dim0_stride lo32
  g1[6] = 0;
  g1[7] = 0;
  tdm_load(g0, g1);
}
#endif

// ---------------------------------------------------------------------------
// f32 -> f16 conversion (plain, and transposing for weights)
// ---------------------------------------------------------------------------
__global__ void f32_to_f16_kernel(const float* __restrict__ src,
                                  _Float16* __restrict__ dst, size_t n) {
  size_t i = (size_t)blockIdx.x * blockDim.x + threadIdx.x;
  if (i < n) dst[i] = (_Float16)src[i];
}

// src: K x N (row-major f32)  ->  dst: N x K (row-major f16)
__global__ void f32_to_f16_t_kernel(const float* __restrict__ src,
                                    _Float16* __restrict__ dst, int K, int N) {
  size_t idx = (size_t)blockIdx.x * blockDim.x + threadIdx.x;
  size_t total = (size_t)K * N;
  if (idx >= total) return;
  size_t n = idx / (size_t)K, k = idx % (size_t)K;
  dst[idx] = (_Float16)src[k * (size_t)N + n];
}

// ---------------------------------------------------------------------------
// f16 GEMM: C[M,N] = A[M,K] @ Bt[N,K]^T, f32 accumulate.
// Block tile 128x128, K-step 32, double-buffered TDM staging of both tiles.
// 8 waves in a 4x2 grid; each computes a 32x64 sub-tile (2x4 WMMA accums).
// ---------------------------------------------------------------------------
__global__ __launch_bounds__(256) void gemm_f16_kernel(
    const _Float16* __restrict__ A, const _Float16* __restrict__ Bt,
    _Float16* __restrict__ C16, _Float16* __restrict__ C16t,
    float* __restrict__ C32, int M, int N, int K) {
  __shared__ __align__(16) _Float16 As[2][128][40];   // [m][k], pad 32->40
  __shared__ __align__(16) _Float16 Bs[2][128][40];   // [n][k], pad 32->40
  const int tid = threadIdx.x;
  const int lane = tid & 31, wave = tid >> 5;
  const int wm = wave & 3, wn = wave >> 2;
  const int lr = lane & 15, hi = lane >> 4;
  const int m0 = blockIdx.y * 128, n0 = blockIdx.x * 128;

  v8f acc[2][4];
  {
    v8f z = {};
    for (int i = 0; i < 2; ++i)
      for (int f = 0; f < 4; ++f) acc[i][f] = z;
  }

  auto compute_tile = [&](int cur) {
    v16h afrag[2], bfrag[4];
    #pragma unroll
    for (int i = 0; i < 2; ++i) {
      int row = wm * 32 + i * 16 + lr;
      v8h lo  = *(const v8h*)(&As[cur][row][hi * 8]);
      v8h hi8 = *(const v8h*)(&As[cur][row][16 + hi * 8]);
      #pragma unroll
      for (int j = 0; j < 8; ++j) { afrag[i][j] = lo[j]; afrag[i][j + 8] = hi8[j]; }
    }
    #pragma unroll
    for (int f = 0; f < 4; ++f) {
      int col = wn * 64 + f * 16 + lr;
      v8h lo  = *(const v8h*)(&Bs[cur][col][hi * 8]);
      v8h hi8 = *(const v8h*)(&Bs[cur][col][16 + hi * 8]);
      #pragma unroll
      for (int j = 0; j < 8; ++j) { bfrag[f][j] = lo[j]; bfrag[f][j + 8] = hi8[j]; }
    }
    #pragma unroll
    for (int i = 0; i < 2; ++i)
      #pragma unroll
      for (int f = 0; f < 4; ++f)
        acc[i][f] = WMMA_F32_F16(afrag[i], bfrag[f], acc[i][f]);
  };

#if defined(HAVE_TDM)
  const int nk = K / 32;
  if (wave == 0) {
    tdm_issue_tile(A,  K, M, m0, 0, &As[0][0][0]);
    tdm_issue_tile(Bt, K, N, n0, 0, &Bs[0][0][0]);
  }
  for (int t = 0; t < nk; ++t) {
    int cur = t & 1;
    if (wave == 0) {
      if (t + 1 < nk) {
        tdm_issue_tile(A,  K, M, m0, (t + 1) * 32, &As[cur ^ 1][0][0]);
        tdm_issue_tile(Bt, K, N, n0, (t + 1) * 32, &Bs[cur ^ 1][0][0]);
        wait_tensor2();   // tile t's two descriptors retired (in-order)
      } else {
        wait_tensor0();
      }
    }
    __syncthreads();
    compute_tile(cur);
    __syncthreads();
  }
#else
  for (int k0 = 0; k0 < K; k0 += 32) {
    #pragma unroll
    for (int c = 0; c < 2; ++c) {
      int idx = tid + c * 256;
      int r = idx >> 2, cc = (idx & 3) << 3;
      v8h av = {};
      if (m0 + r < M) av = *(const v8h*)(A + (size_t)(m0 + r) * K + (k0 + cc));
      *(v8h*)(&As[0][r][cc]) = av;
      v8h bv = {};
      if (n0 + r < N) bv = *(const v8h*)(Bt + (size_t)(n0 + r) * K + (k0 + cc));
      *(v8h*)(&Bs[0][r][cc]) = bv;
    }
    __syncthreads();
    compute_tile(0);
    __syncthreads();
  }
#endif

  // Epilogue: C layout row = j + 8*hi, col = lane%16.
  #pragma unroll
  for (int i = 0; i < 2; ++i) {
    #pragma unroll
    for (int f = 0; f < 4; ++f) {
      int col = n0 + wn * 64 + f * 16 + lr;
      if (col >= N) continue;
      #pragma unroll
      for (int j = 0; j < 8; ++j) {
        int row = m0 + wm * 32 + i * 16 + j + 8 * hi;
        if (row >= M) continue;
        if (C16)  C16[(size_t)row * N + col] = (_Float16)acc[i][f][j];
        if (C16t) C16t[(size_t)col * M + row] = (_Float16)acc[i][f][j];
        if (C32)  C32[(size_t)row * N + col] = acc[i][f][j];
      }
    }
  }
}

// ---------------------------------------------------------------------------
// Build concatenated q/k heads (B,T,H,96): [0,64) compressed, [64,96) rotary
// with (cos+sin) RoPE scale applied on the fly.
// ---------------------------------------------------------------------------
__global__ void build_qk_kernel(const _Float16* __restrict__ qC,
                                const _Float16* __restrict__ qR,
                                const _Float16* __restrict__ kC,
                                const _Float16* __restrict__ kR,
                                _Float16* __restrict__ qh,
                                _Float16* __restrict__ kh,
                                int B, int T, int H) {
  size_t idx = (size_t)blockIdx.x * blockDim.x + threadIdx.x;
  size_t total = (size_t)B * T * H * 96;
  if (idx >= total) return;
  int d = (int)(idx % 96);
  size_t r = idx / 96;
  int h = (int)(r % H);
  size_t bt = r / H;
  int t = (int)(bt % (size_t)T);
  _Float16 qv, kv;
  if (d < 64) {
    qv = qC[bt * 1024 + h * 64 + d];
    kv = kC[bt * 1024 + h * 64 + d];
  } else {
    int j = d - 64;
    int i = j & 15;                       // emb = concat(freqs, freqs)
    // inv = 10000^(-i/16); ln(10000)/16 = 0.5756462732485115
    float ang = (float)t * __expf(-(float)i * 0.57564627324851150f);
    float sc = __cosf(ang) + __sinf(ang);
    qv = (_Float16)((float)qR[bt * 512 + h * 32 + j] * sc);
    kv = (_Float16)((float)kR[bt * 32 + j] * sc);
  }
  qh[idx] = qv;
  kh[idx] = kv;
}

// ---------------------------------------------------------------------------
// Flash attention (causal) with WMMA. Grid: (T/128, B*H).
// Wave owns 16 query rows; streams 64-key K/V tiles, double-buffered with
// async-to-LDS copies (5 per thread per tile -> s_wait_asynccnt 5 retires
// the current tile after the next has been issued). V comes pre-transposed.
// ---------------------------------------------------------------------------
__global__ __launch_bounds__(256) void mla_attn_kernel(
    const _Float16* __restrict__ qh, const _Float16* __restrict__ kh,
    const _Float16* __restrict__ vT, _Float16* __restrict__ yh,
    int B, int T, int H) {
  __shared__ __align__(16) _Float16 Kt[2][64][104];   // [key][d], d = 0..95
  __shared__ __align__(16) _Float16 Vt[2][64][72];    // [d][key]
  __shared__ __align__(16) _Float16 Ps[8][16][72];    // per-wave P scratch
  const int bh = blockIdx.y;
  const int b = bh / H, h = bh % H;
  const int q0 = blockIdx.x * 128;
  const int tid = threadIdx.x, lane = tid & 31, wave = tid >> 5;
  const int lr = lane & 15, hi = lane >> 4;
  const int wr0 = q0 + wave * 16;
  const size_t BT = (size_t)B * T;

  // Resident q A-fragments for the 3 d-chunks (96 = 3*32).
  v16h qf[3];
  {
    const _Float16* qb = qh + (((size_t)b * T + (wr0 + lr)) * H + h) * 96;
    #pragma unroll
    for (int c = 0; c < 3; ++c) {
      v8h lo  = *(const v8h*)(qb + c * 32 + hi * 8);
      v8h hi8 = *(const v8h*)(qb + c * 32 + 16 + hi * 8);
      #pragma unroll
      for (int j = 0; j < 8; ++j) { qf[c][j] = lo[j]; qf[c][j + 8] = hi8[j]; }
    }
  }

  v8f Oacc[4];
  {
    v8f z = {};
    #pragma unroll
    for (int f = 0; f < 4; ++f) Oacc[f] = z;
  }
  float mrow[8], lrow[8];
  #pragma unroll
  for (int j = 0; j < 8; ++j) { mrow[j] = -1e30f; lrow[j] = 0.0f; }

  const float sscale = 0.10206207261596575f;  // 1/sqrt(96)
  const int kend = (q0 + 128 < T) ? (q0 + 128) : T;
  const int ntiles = kend / 64;

  auto stage = [&](int ti, int buf) {
    int kb0 = ti * 64;
#if defined(HAVE_ASYNC)
    for (int c = tid; c < 768; c += 256) {        // K tile 64x96 halves
      int r = c / 12, cc = (c % 12) * 8;
      async_copy_b128(kh + (((size_t)b * T + (kb0 + r)) * H + h) * 96 + cc,
                      &Kt[buf][r][cc]);
    }
    for (int c = tid; c < 512; c += 256) {        // V^T tile 64x64 halves
      int r = c >> 3, ck = (c & 7) << 3;
      async_copy_b128(vT + (size_t)(h * 64 + r) * BT + ((size_t)b * T + kb0 + ck),
                      &Vt[buf][r][ck]);
    }
#else
    for (int c = tid; c < 768; c += 256) {
      int r = c / 12, cc = (c % 12) * 8;
      v8h val = *(const v8h*)(kh + (((size_t)b * T + (kb0 + r)) * H + h) * 96 + cc);
      *(v8h*)(&Kt[buf][r][cc]) = val;
    }
    for (int c = tid; c < 512; c += 256) {
      int r = c >> 3, ck = (c & 7) << 3;
      v8h val = *(const v8h*)(vT + (size_t)(h * 64 + r) * BT +
                              ((size_t)b * T + kb0 + ck));
      *(v8h*)(&Vt[buf][r][ck]) = val;
    }
#endif
  };

  stage(0, 0);
  for (int ti = 0; ti < ntiles; ++ti) {
    int cur = ti & 1;
    if (ti + 1 < ntiles) {
      stage(ti + 1, cur ^ 1);
#if defined(HAVE_ASYNC)
      wait_async5();    // current tile's 5 copies retired (in-order)
#endif
    } else {
#if defined(HAVE_ASYNC)
      wait_async0();
#endif
    }
    __syncthreads();

    int kb0 = ti * 64;
    if (kb0 <= wr0 + 15) {  // wave-uniform: EXEC stays all-ones around WMMA
      v8f S[4];
      #pragma unroll
      for (int kb = 0; kb < 4; ++kb) {
        v8f s = {};
        #pragma unroll
        for (int c = 0; c < 3; ++c) {
          v16h bf;
          v8h lo  = *(const v8h*)(&Kt[cur][kb * 16 + lr][c * 32 + hi * 8]);
          v8h hi8 = *(const v8h*)(&Kt[cur][kb * 16 + lr][c * 32 + 16 + hi * 8]);
          #pragma unroll
          for (int j = 0; j < 8; ++j) { bf[j] = lo[j]; bf[j + 8] = hi8[j]; }
          s = WMMA_F32_F16(qf[c], bf, s);
        }
        int colg = kb0 + kb * 16 + lr;
        #pragma unroll
        for (int j = 0; j < 8; ++j) {
          float v = s[j] * sscale;
          int rowg = wr0 + j + 8 * hi;
          S[kb][j] = (colg > rowg) ? -1e30f : v;
        }
      }
      float mnew[8], alpha[8];
      #pragma unroll
      for (int j = 0; j < 8; ++j) {
        float v = fmaxf(fmaxf(S[0][j], S[1][j]), fmaxf(S[2][j], S[3][j]));
        #pragma unroll
        for (int off = 1; off < 16; off <<= 1) v = fmaxf(v, __shfl_xor(v, off, 16));
        mnew[j] = fmaxf(mrow[j], v);
        alpha[j] = __expf(mrow[j] - mnew[j]);
      }
      float rsum[8];
      #pragma unroll
      for (int j = 0; j < 8; ++j) rsum[j] = 0.0f;
      #pragma unroll
      for (int kb = 0; kb < 4; ++kb) {
        #pragma unroll
        for (int j = 0; j < 8; ++j) {
          float p = __expf(S[kb][j] - mnew[j]);
          rsum[j] += p;
          Ps[wave][j + 8 * hi][kb * 16 + lr] = (_Float16)p;
        }
      }
      #pragma unroll
      for (int j = 0; j < 8; ++j) {
        float v = rsum[j];
        #pragma unroll
        for (int off = 1; off < 16; off <<= 1) v += __shfl_xor(v, off, 16);
        lrow[j] = lrow[j] * alpha[j] + v;
        mrow[j] = mnew[j];
      }
      #pragma unroll
      for (int f = 0; f < 4; ++f)
        #pragma unroll
        for (int j = 0; j < 8; ++j) Oacc[f][j] = Oacc[f][j] * alpha[j];
      // O += P @ V (keys in 2 chunks of 32).
      #pragma unroll
      for (int kc = 0; kc < 2; ++kc) {
        v16h af;
        v8h lo  = *(const v8h*)(&Ps[wave][lr][kc * 32 + hi * 8]);
        v8h hi8 = *(const v8h*)(&Ps[wave][lr][kc * 32 + 16 + hi * 8]);
        #pragma unroll
        for (int j = 0; j < 8; ++j) { af[j] = lo[j]; af[j + 8] = hi8[j]; }
        #pragma unroll
        for (int f = 0; f < 4; ++f) {
          v16h bf;
          v8h blo = *(const v8h*)(&Vt[cur][f * 16 + lr][kc * 32 + hi * 8]);
          v8h bhi = *(const v8h*)(&Vt[cur][f * 16 + lr][kc * 32 + 16 + hi * 8]);
          #pragma unroll
          for (int j = 0; j < 8; ++j) { bf[j] = blo[j]; bf[j + 8] = bhi[j]; }
          Oacc[f] = WMMA_F32_F16(af, bf, Oacc[f]);
        }
      }
    }
    __syncthreads();
  }

  // Normalize and write y (B,T,H,64) f16.
  #pragma unroll
  for (int f = 0; f < 4; ++f) {
    #pragma unroll
    for (int j = 0; j < 8; ++j) {
      int rowg = wr0 + j + 8 * hi;
      float v = Oacc[f][j] / lrow[j];
      yh[(((size_t)b * T + rowg) * H + h) * 64 + f * 16 + lr] = (_Float16)v;
    }
  }
}

// ---------------------------------------------------------------------------
// Launch
// ---------------------------------------------------------------------------
extern "C" void kernel_launch(void* const* d_in, const int* in_sizes, int n_in,
                              void* d_out, int out_size, void* d_ws, size_t ws_size,
                              hipStream_t stream) {
  const int B = 4, T = 2048, C = 1024, H = 16;
  const int DCH = 4096, DC = 256, DHR = 32;
  const size_t BT = (size_t)B * T;  // 8192

  _Float16* w = (_Float16*)d_ws;
  size_t off = 0;
  auto alloc = [&](size_t n) { _Float16* p = w + off; off += n; return p; };

  _Float16* xh   = alloc(BT * C);
  _Float16* WDQ  = alloc((size_t)C * DCH);     // stored transposed: [N][K]
  _Float16* WUQ  = alloc((size_t)DCH * C);
  _Float16* WDKV = alloc((size_t)C * DC);
  _Float16* WUK  = alloc((size_t)DC * C);
  _Float16* WUV  = alloc((size_t)DC * C);
  _Float16* WQR  = alloc((size_t)DCH * (DHR * H));
  _Float16* WKR  = alloc((size_t)C * DHR);
  _Float16* WO   = alloc((size_t)C * C);
  _Float16* cQ   = alloc(BT * DCH);
  _Float16* qC   = alloc(BT * C);
  _Float16* cKV  = alloc(BT * DC);
  _Float16* kC   = alloc(BT * C);
  _Float16* vT   = alloc(BT * C);              // v stored transposed [C][BT]
  _Float16* qR   = alloc(BT * (size_t)(DHR * H));
  _Float16* kR   = alloc(BT * DHR);
  _Float16* qhB  = alloc(BT * (size_t)H * 96);
  _Float16* khB  = alloc(BT * (size_t)H * 96);
  _Float16* yh   = alloc(BT * C);

  // x: plain convert
  {
    size_t n = (size_t)in_sizes[0];
    int blocks = (int)((n + 255) / 256);
    f32_to_f16_kernel<<<blocks, 256, 0, stream>>>((const float*)d_in[0], xh, n);
  }
  // weights: convert + transpose (K x N f32 -> N x K f16)
  struct WInfo { _Float16* dst; int k, n; };
  const WInfo wi[8] = {
    { WDQ,  C,   DCH },     { WUQ,  DCH, C },
    { WDKV, C,   DC },      { WUK,  DC,  C },
    { WUV,  DC,  C },       { WQR,  DCH, DHR * H },
    { WKR,  C,   DHR },     { WO,   C,   C },
  };
  for (int i = 0; i < 8; ++i) {
    size_t n = (size_t)wi[i].k * wi[i].n;
    int blocks = (int)((n + 255) / 256);
    f32_to_f16_t_kernel<<<blocks, 256, 0, stream>>>(
        (const float*)d_in[i + 1], wi[i].dst, wi[i].k, wi[i].n);
  }

  dim3 blk(256);
  auto gemm = [&](const _Float16* A, const _Float16* Bt, _Float16* C16,
                  _Float16* C16t, float* C32, int M, int N, int K) {
    dim3 g((N + 127) / 128, M / 128);
    gemm_f16_kernel<<<g, blk, 0, stream>>>(A, Bt, C16, C16t, C32, M, N, K);
  };

  gemm(xh,  WDQ,  cQ,  nullptr, nullptr, (int)BT, DCH, C);        // c_Q
  gemm(cQ,  WUQ,  qC,  nullptr, nullptr, (int)BT, C,   DCH);      // q_C
  gemm(xh,  WDKV, cKV, nullptr, nullptr, (int)BT, DC,  C);        // c_KV
  gemm(cKV, WUK,  kC,  nullptr, nullptr, (int)BT, C,   DC);       // k_C
  gemm(cKV, WUV,  nullptr, vT,  nullptr, (int)BT, C,   DC);       // v (transposed)
  gemm(cQ,  WQR,  qR,  nullptr, nullptr, (int)BT, DHR * H, DCH);  // q_R raw
  gemm(xh,  WKR,  kR,  nullptr, nullptr, (int)BT, DHR, C);        // k_R raw

  {
    size_t total = BT * (size_t)H * 96;
    int blocks = (int)((total + 255) / 256);
    build_qk_kernel<<<blocks, 256, 0, stream>>>(qC, qR, kC, kR, qhB, khB, B, T, H);
  }
  {
    dim3 g(T / 128, B * H);
    mla_attn_kernel<<<g, blk, 0, stream>>>(qhB, khB, vT, yh, B, T, H);
  }
  gemm(yh, WO, nullptr, nullptr, (float*)d_out, (int)BT, C, C);   // out projection

  (void)n_in; (void)out_size; (void)ws_size;
}
